// GATGCNGRU_75118978007589
// MI455X (gfx1250) — matrix operand order repaired and checked by reading, not compiled, not run on hardware
//
#include <hip/hip_runtime.h>
#include <stdint.h>

#define Nn 20000
#define Fd 128
#define Hd 256
#define Tt 12
#define Ee 320000
#define Et (Ee + Nn)
#define SLOPE 0.2f

typedef __attribute__((ext_vector_type(16))) __bf16 v16bf;
typedef __attribute__((ext_vector_type(8)))  __bf16 v8bf;
typedef __attribute__((ext_vector_type(8)))  float  v8f;

__device__ __forceinline__ __bf16 f2bf(float f) {
  unsigned u = __float_as_uint(f);
  unsigned r = (u + 0x7fffu + ((u >> 16) & 1u)) >> 16;
  unsigned short s = (unsigned short)r;
  __bf16 o;
  __builtin_memcpy(&o, &s, 2);
  return o;
}
__device__ __forceinline__ float sigmoidf_(float x) { return 1.f / (1.f + __expf(-x)); }
__device__ __forceinline__ float keydec(unsigned k) {
  unsigned b = (k & 0x80000000u) ? (k & 0x7fffffffu) : ~k;
  return __uint_as_float(b);
}

// ---------------- fills ----------------
__global__ void k_fill_f32(float* __restrict__ p, float v, int n) {
  int i = blockIdx.x * blockDim.x + threadIdx.x;
  if (i < n) p[i] = v;
}
__global__ void k_fill_u32(unsigned* __restrict__ p, unsigned v, int n) {
  int i = blockIdx.x * blockDim.x + threadIdx.x;
  if (i < n) p[i] = v;
}
__global__ void k_copy_f32(const float* __restrict__ in, float* __restrict__ out, int n) {
  int i = blockIdx.x * blockDim.x + threadIdx.x;
  if (i < n) out[i] = in[i];
}
__global__ void k_cvt_bf16(const float* __restrict__ in, __bf16* __restrict__ out, int n) {
  int i = blockIdx.x * blockDim.x + threadIdx.x;
  if (i < n) out[i] = f2bf(in[i]);
}

// ---------------- weight packers (f32 -> bf16, transposed: Bt[n][k]) ----------------
__global__ void k_build_lr(const float* __restrict__ Wl, const float* __restrict__ Wr,
                           __bf16* __restrict__ Bt) {
  int i = blockIdx.x * blockDim.x + threadIdx.x;     // 2*Fd*Fd
  if (i >= 2 * Fd * Fd) return;
  int k = i % Fd;
  int n = (i / Fd) % Fd;
  int g = i / (Fd * Fd);
  const float* W = g ? Wr : Wl;
  Bt[i] = f2bf(W[k * Fd + n]);
}
__global__ void k_build_x(const float* __restrict__ Wz, const float* __restrict__ Wr_,
                          const float* __restrict__ Wh, __bf16* __restrict__ Bt) {
  int i = blockIdx.x * blockDim.x + threadIdx.x;     // 3*Hd * 3*Fd
  if (i >= 3 * Hd * 3 * Fd) return;
  int col = i % (3 * Fd);
  int rn  = i / (3 * Fd);
  int g = rn / Hd, n = rn % Hd;
  int kk = col / Fd, f = col % Fd;
  const float* W = (g == 0) ? Wz : ((g == 1) ? Wr_ : Wh);
  Bt[i] = f2bf(W[(kk * Fd + f) * Hd + n]);
}
__global__ void k_build_h2(const float* __restrict__ Wz, const float* __restrict__ Wr_,
                           __bf16* __restrict__ Bt) {
  int i = blockIdx.x * blockDim.x + threadIdx.x;     // 2*Hd * 3*Hd
  if (i >= 2 * Hd * 3 * Hd) return;
  int col = i % (3 * Hd);
  int rn  = i / (3 * Hd);
  int g = rn / Hd, n = rn % Hd;
  int kk = col / Hd, hh = col % Hd;
  const float* W = g ? Wr_ : Wz;
  Bt[i] = f2bf(W[(kk * Hd + hh) * Hd + n]);
}
__global__ void k_build_hh(const float* __restrict__ Wh, __bf16* __restrict__ Bt) {
  int i = blockIdx.x * blockDim.x + threadIdx.x;     // Hd * 3*Hd
  if (i >= Hd * 3 * Hd) return;
  int col = i % (3 * Hd);
  int n   = i / (3 * Hd);
  int kk = col / Hd, hh = col % Hd;
  Bt[i] = f2bf(Wh[(kk * Hd + hh) * Hd + n]);
}

// ---------------- WMMA bf16 GEMM: C[M x Nc] = A[M x Kd] @ Bt^T, Bt is [Nc x Kd] ----------------
// grid: (Nc/64, ceil(M/128)); block 256 = 8 waves; wave -> 16(M) x 64(N) strip.
__global__ __launch_bounds__(256) void k_gemm_bf16(
    const __bf16* __restrict__ A, int lda,
    const __bf16* __restrict__ Bt, int Kd,
    float* __restrict__ C, int ldc, int M) {
  const int lane = threadIdx.x & 31;
  const int wave = threadIdx.x >> 5;
  const int m0 = blockIdx.y * 128 + wave * 16;
  const int n0 = blockIdx.x * 64;
  int row = m0 + (lane & 15);
  if (row >= M) row = M - 1;                 // clamp for loads; store is guarded
  const int ka   = (lane >> 4) * 16;         // A fragment half: K +0 / +16
  const int kb16 = (lane >> 4) * 16;         // B fragment half: K +0 / +16
  const __bf16* ap = A + (size_t)row * lda + ka;
  const __bf16* b0 = Bt + (size_t)(n0 + 0  + (lane & 15)) * Kd + kb16;
  const __bf16* b1 = b0 + (size_t)16 * Kd;
  const __bf16* b2 = b0 + (size_t)32 * Kd;
  const __bf16* b3 = b0 + (size_t)48 * Kd;
  v8f acc0 = {}, acc1 = {}, acc2 = {}, acc3 = {};
  for (int kk = 0; kk < Kd; kk += 32) {
    __builtin_prefetch(ap + kk + 128, 0, 0);
    v8bf alo = *(const v8bf*)(ap + kk);
    v8bf ahi = *(const v8bf*)(ap + kk + 8);
    v16bf a = __builtin_shufflevector(alo, ahi, 0, 1, 2, 3, 4, 5, 6, 7,
                                      8, 9, 10, 11, 12, 13, 14, 15);
    v16bf vb0 = *(const v16bf*)(b0 + kk);
    v16bf vb1 = *(const v16bf*)(b1 + kk);
    v16bf vb2 = *(const v16bf*)(b2 + kk);
    v16bf vb3 = *(const v16bf*)(b3 + kk);
    acc0 = __builtin_amdgcn_wmma_f32_16x16x32_bf16(false, a, false, vb0, (short)0, acc0, false, false);
    acc1 = __builtin_amdgcn_wmma_f32_16x16x32_bf16(false, a, false, vb1, (short)0, acc1, false, false);
    acc2 = __builtin_amdgcn_wmma_f32_16x16x32_bf16(false, a, false, vb2, (short)0, acc2, false, false);
    acc3 = __builtin_amdgcn_wmma_f32_16x16x32_bf16(false, a, false, vb3, (short)0, acc3, false, false);
  }
  const int mh = m0 + (lane >> 4) * 8;
  const int nc = n0 + (lane & 15);
#pragma unroll
  for (int r = 0; r < 8; ++r) {
    int mr = mh + r;
    if (mr < M) {
      float* c = C + (size_t)mr * ldc + nc;
      c[0]  = acc0[r];
      c[16] = acc1[r];
      c[32] = acc2[r];
      c[48] = acc3[r];
    }
  }
}

// ---------------- GATv2 attention ----------------
// one wave per edge; XLR row = [xl(128) | xr(128)]
__global__ void k_edge_score(const float* __restrict__ XLR, const int* __restrict__ ei,
                             const float* __restrict__ bl, const float* __restrict__ br,
                             const float* __restrict__ att, float* __restrict__ e) {
  int lane = threadIdx.x & 31;
  int ed = blockIdx.x * (blockDim.x >> 5) + (threadIdx.x >> 5);
  if (ed >= Et) return;
  int s, d;
  if (ed < Ee) { s = ei[ed]; d = ei[Ee + ed]; } else { s = d = ed - Ee; }
  const float* xl = XLR + (size_t)s * 256;
  const float* xr = XLR + (size_t)d * 256 + 128;
  float acc = 0.f;
  for (int f = lane; f < Fd; f += 32) {
    float v = xl[f] + bl[f] + xr[f] + br[f];
    v = (v > 0.f) ? v : v * SLOPE;
    acc += att[f] * v;
  }
  for (int o = 16; o; o >>= 1) acc += __shfl_xor(acc, o, 32);
  if (!lane) e[ed] = acc;
}
__global__ void k_segmax(const float* __restrict__ e, const int* __restrict__ ei,
                         unsigned* __restrict__ keys) {
  int i = blockIdx.x * blockDim.x + threadIdx.x;
  if (i >= Et) return;
  int d = (i < Ee) ? ei[Ee + i] : (i - Ee);
  unsigned b = __float_as_uint(e[i]);
  unsigned key = (b & 0x80000000u) ? ~b : (b | 0x80000000u);
  atomicMax(&keys[d], key);
}
__global__ void k_expsum(const float* __restrict__ e, const int* __restrict__ ei,
                         const unsigned* __restrict__ keys, float* __restrict__ ee,
                         float* __restrict__ esum) {
  int i = blockIdx.x * blockDim.x + threadIdx.x;
  if (i >= Et) return;
  int d = (i < Ee) ? ei[Ee + i] : (i - Ee);
  float v = __expf(e[i] - keydec(keys[d]));
  ee[i] = v;
  atomicAdd(&esum[d], v);
}
__global__ void k_alpha(const float* __restrict__ ee, const int* __restrict__ ei,
                        const float* __restrict__ esum, float* __restrict__ alpha,
                        float* __restrict__ deg) {
  int i = blockIdx.x * blockDim.x + threadIdx.x;
  if (i >= Et) return;
  int s, d;
  if (i < Ee) { s = ei[i]; d = ei[Ee + i]; } else { s = d = i - Ee; }
  float a = ee[i] / esum[d];
  alpha[i] = a;
  atomicAdd(&deg[s], a);
}
__global__ void k_dis(const float* __restrict__ deg, float* __restrict__ dis) {
  int i = blockIdx.x * blockDim.x + threadIdx.x;
  if (i >= Nn) return;
  float dg = deg[i];
  dis[i] = (dg > 0.f) ? rsqrtf(dg) : 0.f;
}
__global__ void k_lw(const float* __restrict__ alpha, const int* __restrict__ ei,
                     const float* __restrict__ dis, float* __restrict__ lw) {
  int i = blockIdx.x * blockDim.x + threadIdx.x;
  if (i >= Et) return;
  int s, d;
  if (i < Ee) { s = ei[i]; d = ei[Ee + i]; } else { s = d = i - Ee; }
  lw[i] = -dis[s] * alpha[i] * dis[d];
}

// ---------------- sparse propagation: Yout[dst] += lw * Yin[src] ----------------
__global__ void k_scatter(const float* __restrict__ Yin, int ldin,
                          float* __restrict__ Yout, int logW,
                          const int* __restrict__ ei, const float* __restrict__ lw) {
  int i = blockIdx.x * blockDim.x + threadIdx.x;
  int W = 1 << logW;
  int total = Et << logW;
  if (i >= total) return;
  int ed = i >> logW;
  int f = i & (W - 1);
  int s, d;
  if (ed < Ee) { s = ei[ed]; d = ei[Ee + ed]; } else { s = d = ed - Ee; }
  atomicAdd(&Yout[(size_t)d * W + f], lw[ed] * Yin[(size_t)s * ldin + f]);
}

// pack [Tx0 | Tx1 | 2*P2 - Tx0] -> bf16, row stride 3W
__global__ void k_pack3(const float* __restrict__ t0, int ld0,
                        const float* __restrict__ t1, const float* __restrict__ p2,
                        __bf16* __restrict__ out, int logW) {
  int i = blockIdx.x * blockDim.x + threadIdx.x;
  int W = 1 << logW;
  int total = Nn << logW;
  if (i >= total) return;
  int n = i >> logW;
  int w = i & (W - 1);
  float a = t0[(size_t)n * ld0 + w];
  __bf16* o = out + (size_t)n * 3 * W;
  o[w]         = f2bf(a);
  o[W + w]     = f2bf(t1[(size_t)n * W + w]);
  o[2 * W + w] = f2bf(2.f * p2[(size_t)n * W + w] - a);
}

// ---------------- GRU elementwise ----------------
__global__ void k_zr(const float* __restrict__ Gx, const float* __restrict__ Gh,
                     const float* __restrict__ bxz, const float* __restrict__ bhz,
                     const float* __restrict__ bxr, const float* __restrict__ bhr,
                     const float* __restrict__ h, float* __restrict__ Z,
                     float* __restrict__ HR) {
  int i = blockIdx.x * blockDim.x + threadIdx.x;
  if (i >= Nn * Hd) return;
  int n = i >> 8, w = i & 255;
  float z = sigmoidf_(Gx[(size_t)n * 768 + w]       + bxz[w] + Gh[(size_t)n * 512 + w]       + bhz[w]);
  float r = sigmoidf_(Gx[(size_t)n * 768 + 256 + w] + bxr[w] + Gh[(size_t)n * 512 + 256 + w] + bhr[w]);
  Z[i] = z;
  HR[i] = h[i] * r;
}
__global__ void k_hupd(const float* __restrict__ Gx, const float* __restrict__ Ghh,
                       const float* __restrict__ bxh, const float* __restrict__ bhh,
                       const float* __restrict__ Z, float* __restrict__ h) {
  int i = blockIdx.x * blockDim.x + threadIdx.x;
  if (i >= Nn * Hd) return;
  int n = i >> 8, w = i & 255;
  float ht = tanhf(Gx[(size_t)n * 768 + 512 + w] + bxh[w] + Ghh[i] + bhh[w]);
  float z = Z[i];
  h[i] = z * h[i] + (1.f - z) * ht;
}

// ---------------- readout ----------------
__global__ void k_readout(const float* __restrict__ h, const float* __restrict__ Wlin,
                          const float* __restrict__ blin, float* __restrict__ out) {
  int lane = threadIdx.x & 31;
  int node = blockIdx.x * (blockDim.x >> 5) + (threadIdx.x >> 5);
  if (node >= Nn) return;
  const float* hp = h + (size_t)node * Hd;
  float acc = 0.f;
  for (int w = lane; w < Hd; w += 32) acc += hp[w] * Wlin[w];
  for (int o = 16; o; o >>= 1) acc += __shfl_xor(acc, o, 32);
  if (!lane) out[node] = acc + blin[0];
}

// ---------------- host ----------------
static inline int cdiv(long a, long b) { return (int)((a + b - 1) / b); }

extern "C" void kernel_launch(void* const* d_in, const int* in_sizes, int n_in,
                              void* d_out, int out_size, void* d_ws, size_t ws_size,
                              hipStream_t stream) {
  const float* x          = (const float*)d_in[0];   // [T,N,F]
  const int*   edge_index = (const int*)d_in[1];     // [T,2,E]
  const float* Wl  = (const float*)d_in[3];
  const float* bl  = (const float*)d_in[4];
  const float* Wr  = (const float*)d_in[5];
  const float* br  = (const float*)d_in[6];
  const float* att = (const float*)d_in[7];
  const float* Wxz = (const float*)d_in[8];
  const float* Whz = (const float*)d_in[9];
  const float* bxz = (const float*)d_in[10];
  const float* bhz = (const float*)d_in[11];
  const float* Wxr = (const float*)d_in[12];
  const float* Whr = (const float*)d_in[13];
  const float* bxr = (const float*)d_in[14];
  const float* bhr = (const float*)d_in[15];
  const float* Wxh = (const float*)d_in[16];
  const float* Whh = (const float*)d_in[17];
  const float* bxh = (const float*)d_in[18];
  const float* bhh = (const float*)d_in[19];
  const float* Wlin = (const float*)d_in[20];
  const float* blin = (const float*)d_in[21];

  // bump allocator over d_ws
  char* wsp = (char*)d_ws;
  auto alloc = [&](size_t bytes) -> void* {
    void* r = (void*)wsp;
    wsp += (bytes + 255) & ~(size_t)255;
    return r;
  };
  float*    Hbuf  = (float*)alloc((size_t)Nn * Hd * 4);
  float*    XLR   = (float*)alloc((size_t)Nn * 256 * 4);
  float*    esc   = (float*)alloc((size_t)Et * 4);
  float*    eex   = (float*)alloc((size_t)Et * 4);
  float*    alpha = (float*)alloc((size_t)Et * 4);
  float*    lwb   = (float*)alloc((size_t)Et * 4);
  unsigned* keys  = (unsigned*)alloc((size_t)Nn * 4);
  float*    esum  = (float*)alloc((size_t)Nn * 4);
  float*    deg   = (float*)alloc((size_t)Nn * 4);
  float*    dis   = (float*)alloc((size_t)Nn * 4);
  float*    P1    = (float*)alloc((size_t)Nn * Hd * 4);
  float*    P2    = (float*)alloc((size_t)Nn * Hd * 4);
  float*    Gx    = (float*)alloc((size_t)Nn * 768 * 4);
  float*    Gh    = (float*)alloc((size_t)Nn * 512 * 4);
  float*    Ghh   = (float*)alloc((size_t)Nn * 256 * 4);
  float*    Zbuf  = (float*)alloc((size_t)Nn * 256 * 4);
  float*    HRbuf = (float*)alloc((size_t)Nn * 256 * 4);
  __bf16*   Xa    = (__bf16*)alloc((size_t)Nn * 128 * 2);
  __bf16*   TxXc  = (__bf16*)alloc((size_t)Nn * 384 * 2);
  __bf16*   TxHc  = (__bf16*)alloc((size_t)Nn * 768 * 2);
  __bf16*   TxHRc = (__bf16*)alloc((size_t)Nn * 768 * 2);
  __bf16*   BtLR  = (__bf16*)alloc((size_t)256 * 128 * 2);
  __bf16*   BtX   = (__bf16*)alloc((size_t)768 * 384 * 2);
  __bf16*   BtH   = (__bf16*)alloc((size_t)512 * 768 * 2);
  __bf16*   BtHH  = (__bf16*)alloc((size_t)256 * 768 * 2);

  const int TB = 256;
  const int MG = cdiv(Nn, 128);   // GEMM grid.y

  // one-time weight packing (idempotent per call)
  k_build_lr<<<cdiv(2 * Fd * Fd, TB), TB, 0, stream>>>(Wl, Wr, BtLR);
  k_build_x <<<cdiv(3 * Hd * 3 * Fd, TB), TB, 0, stream>>>(Wxz, Wxr, Wxh, BtX);
  k_build_h2<<<cdiv(2 * Hd * 3 * Hd, TB), TB, 0, stream>>>(Whz, Whr, BtH);
  k_build_hh<<<cdiv(Hd * 3 * Hd, TB), TB, 0, stream>>>(Whh, BtHH);
  k_fill_f32<<<cdiv(Nn * Hd, TB), TB, 0, stream>>>(Hbuf, 0.f, Nn * Hd);

  for (int t = 0; t < Tt; ++t) {
    const float* xt = x + (size_t)t * Nn * Fd;
    const int*   ei = edge_index + (size_t)t * 2 * Ee;

    // --- GATv2 features: XLR = xt @ [Wl | Wr] ---
    k_cvt_bf16<<<cdiv(Nn * Fd, TB), TB, 0, stream>>>(xt, Xa, Nn * Fd);
    k_gemm_bf16<<<dim3(4, MG), TB, 0, stream>>>(Xa, 128, BtLR, 128, XLR, 256, Nn);

    // --- attention softmax over dst + scaled-Laplacian edge weights ---
    k_fill_u32<<<cdiv(Nn, TB), TB, 0, stream>>>(keys, 0u, Nn);
    k_fill_f32<<<cdiv(Nn, TB), TB, 0, stream>>>(esum, 0.f, Nn);
    k_fill_f32<<<cdiv(Nn, TB), TB, 0, stream>>>(deg, 0.f, Nn);
    k_edge_score<<<cdiv(Et, 8), TB, 0, stream>>>(XLR, ei, bl, br, att, esc);
    k_segmax<<<cdiv(Et, TB), TB, 0, stream>>>(esc, ei, keys);
    k_expsum<<<cdiv(Et, TB), TB, 0, stream>>>(esc, ei, keys, eex, esum);
    k_alpha <<<cdiv(Et, TB), TB, 0, stream>>>(eex, ei, esum, alpha, deg);
    k_dis   <<<cdiv(Nn, TB), TB, 0, stream>>>(deg, dis);
    k_lw    <<<cdiv(Et, TB), TB, 0, stream>>>(alpha, ei, dis, lwb);

    // --- Chebyshev basis on x (width 128) ---
    k_fill_f32<<<cdiv(Nn * 128, TB), TB, 0, stream>>>(P1, 0.f, Nn * 128);
    k_scatter<<<cdiv((long)Et * 128, TB), TB, 0, stream>>>(xt, 128, P1, 7, ei, lwb);
    k_fill_f32<<<cdiv(Nn * 128, TB), TB, 0, stream>>>(P2, 0.f, Nn * 128);
    k_scatter<<<cdiv((long)Et * 128, TB), TB, 0, stream>>>(P1, 128, P2, 7, ei, lwb);
    k_pack3<<<cdiv(Nn * 128, TB), TB, 0, stream>>>(xt, 128, P1, P2, TxXc, 7);
    // Gx[:, z|r|h] = [Tx0|Tx1|Tx2] @ BtX^T
    k_gemm_bf16<<<dim3(12, MG), TB, 0, stream>>>(TxXc, 384, BtX, 384, Gx, 768, Nn);

    // --- Chebyshev basis on h (width 256), gates z,r ---
    k_fill_f32<<<cdiv(Nn * 256, TB), TB, 0, stream>>>(P1, 0.f, Nn * 256);
    k_scatter<<<cdiv((long)Et * 256, TB), TB, 0, stream>>>(Hbuf, 256, P1, 8, ei, lwb);
    k_fill_f32<<<cdiv(Nn * 256, TB), TB, 0, stream>>>(P2, 0.f, Nn * 256);
    k_scatter<<<cdiv((long)Et * 256, TB), TB, 0, stream>>>(P1, 256, P2, 8, ei, lwb);
    k_pack3<<<cdiv(Nn * 256, TB), TB, 0, stream>>>(Hbuf, 256, P1, P2, TxHc, 8);
    k_gemm_bf16<<<dim3(8, MG), TB, 0, stream>>>(TxHc, 768, BtH, 768, Gh, 512, Nn);

    // --- gates Z, R; candidate input h*R ---
    k_zr<<<cdiv(Nn * Hd, TB), TB, 0, stream>>>(Gx, Gh, bxz, bhz, bxr, bhr, Hbuf, Zbuf, HRbuf);

    // --- Chebyshev basis on h*R, gate h ---
    k_fill_f32<<<cdiv(Nn * 256, TB), TB, 0, stream>>>(P1, 0.f, Nn * 256);
    k_scatter<<<cdiv((long)Et * 256, TB), TB, 0, stream>>>(HRbuf, 256, P1, 8, ei, lwb);
    k_fill_f32<<<cdiv(Nn * 256, TB), TB, 0, stream>>>(P2, 0.f, Nn * 256);
    k_scatter<<<cdiv((long)Et * 256, TB), TB, 0, stream>>>(P1, 256, P2, 8, ei, lwb);
    k_pack3<<<cdiv(Nn * 256, TB), TB, 0, stream>>>(HRbuf, 256, P1, P2, TxHRc, 8);
    k_gemm_bf16<<<dim3(4, MG), TB, 0, stream>>>(TxHRc, 768, BtHH, 768, Ghh, 256, Nn);

    // --- state update ---
    k_hupd<<<cdiv(Nn * Hd, TB), TB, 0, stream>>>(Gx, Ghh, bxh, bhh, Zbuf, Hbuf);
  }

  // outputs: [out(N) | h(N*H)]
  float* out = (float*)d_out;
  k_readout<<<cdiv(Nn, 8), TB, 0, stream>>>(Hbuf, Wlin, blin, out);
  k_copy_f32<<<cdiv(Nn * Hd, TB), TB, 0, stream>>>(Hbuf, out + Nn, Nn * Hd);
}
